// BDG_ODE_29927332118640
// MI455X (gfx1250) — matrix operand order repaired and checked by reading, not compiled, not run on hardware
//
#include <hip/hip_runtime.h>
#include <math.h>

// Problem constants (match reference)
#define NROW 4096
#define INDIM 64
#define HID 128
#define CH 256      // 2*HID state width
#define TSTEPS 8

typedef __attribute__((ext_vector_type(16))) __bf16       v16bf;
typedef __attribute__((ext_vector_type(8)))  float        v8f;
typedef __attribute__((ext_vector_type(8)))  unsigned int v8u;
typedef __attribute__((ext_vector_type(4)))  unsigned int v4u;

union ABfrag { v8u u; v16bf v; };

// ---------------------------------------------------------------------------
// WMMA GEMM, bf16 split emulation (hi*hi + lo*hi + hi*lo), fp32 accumulation.
//   A row-major [M][K] as Ah+Al.  B stored transposed: BT[n][k] (ld=K).
// Block = 256 threads (8 wave32) -> 128x64 C tile; each wave owns a 32x32
// quadrant (2x2 WMMA tiles) so A/B fragments are each reused twice:
// 12 WMMAs per 16 ds_load_b128 fragment loads per wave per 32-K chunk.
// outMode 0: C fp32 = act(acc + bias)   (ldC = Ndim)
// outMode 1: split epilogue -> Chi/Clo bf16 (feeds the next GEMM's A side)
// ---------------------------------------------------------------------------
__global__ __launch_bounds__(256)
void wmma_gemm_bf16x3(const __bf16* __restrict__ Ah, const __bf16* __restrict__ Al,
                      const __bf16* __restrict__ BhT, const __bf16* __restrict__ BlT,
                      const float* __restrict__ bias, float* __restrict__ C,
                      __bf16* __restrict__ Chi, __bf16* __restrict__ Clo,
                      int M, int Ndim, int Kd, int relu, int outMode)
{
    // padded row stride 40 bf16 (80B) avoids LDS bank conflicts on gathers
    __shared__ __align__(16) __bf16 As[2][128][40];
    __shared__ __align__(16) __bf16 Bs[2][64][40];

    const int tid     = threadIdx.x;
    const int lane    = tid & 31;
    const int wave    = tid >> 5;
    const int waveRow = (wave >> 1) << 5;   // 0,32,64,96
    const int waveCol = (wave & 1) << 5;    // 0,32
    const int mBase   = blockIdx.y << 7;    // 128-row block tile
    const int nBase   = blockIdx.x << 6;    // 64-col block tile
    const int lr      = tid >> 2;           // 0..63
    const int lk      = (tid & 3) << 3;     // 0,8,16,24 (8 bf16 per uint4)
    const int lhalf   = lane >> 4;          // 0/1
    const int l15     = lane & 15;

    v8f acc[2][2] = {};

    for (int k0 = 0; k0 < Kd; k0 += 32) {
        const size_t aoff0 = (size_t)(mBase + lr) * Kd + k0 + lk;
        const size_t aoff1 = aoff0 + (size_t)64 * Kd;
        const size_t boff  = (size_t)(nBase + lr) * Kd + k0 + lk;
        *(v4u*)&As[0][lr][lk]      = *(const v4u*)(Ah  + aoff0);
        *(v4u*)&As[0][lr + 64][lk] = *(const v4u*)(Ah  + aoff1);
        *(v4u*)&As[1][lr][lk]      = *(const v4u*)(Al  + aoff0);
        *(v4u*)&As[1][lr + 64][lk] = *(const v4u*)(Al  + aoff1);
        *(v4u*)&Bs[0][lr][lk]      = *(const v4u*)(BhT + boff);
        *(v4u*)&Bs[1][lr][lk]      = *(const v4u*)(BlT + boff);
        if (k0 + 32 < Kd) {
            __builtin_prefetch(Ah  + aoff0 + 32, 0, 3);
            __builtin_prefetch(Ah  + aoff1 + 32, 0, 3);
            __builtin_prefetch(BhT + boff  + 32, 0, 3);
        }
        __syncthreads();

        ABfrag ah0, al0, ah1, al1, b0h, b0l, b1h, b1l;
        const int mr0 = waveRow + l15;
        const int mr1 = mr0 + 16;
        #pragma unroll
        for (int j = 0; j < 8; ++j) {
            // A-frag (16x32 bf16): dword j covers K = 2*(j&3) + 16*(j>>2) (+8 hi lanes)
            const int ka = ((j & 3) << 1) + ((j >> 2) << 4) + (lhalf << 3);
            ah0.u[j] = *(const unsigned int*)&As[0][mr0][ka];
            al0.u[j] = *(const unsigned int*)&As[1][mr0][ka];
            ah1.u[j] = *(const unsigned int*)&As[0][mr1][ka];
            al1.u[j] = *(const unsigned int*)&As[1][mr1][ka];
        }
        const int n0 = waveCol + l15;
        const int n1 = n0 + 16;
        #pragma unroll
        for (int j = 0; j < 8; ++j) {
            // B-frag (32x16 bf16): dword j covers K = 2*(j&3) + 16*(j>>2) (+8 hi lanes)
            const int kb = ((j & 3) << 1) + ((j >> 2) << 4) + (lhalf << 3);
            b0h.u[j] = *(const unsigned int*)&Bs[0][n0][kb];
            b0l.u[j] = *(const unsigned int*)&Bs[1][n0][kb];
            b1h.u[j] = *(const unsigned int*)&Bs[0][n1][kb];
            b1l.u[j] = *(const unsigned int*)&Bs[1][n1][kb];
        }
        acc[0][0] = __builtin_amdgcn_wmma_f32_16x16x32_bf16(false, ah0.v, false, b0h.v, (short)0, acc[0][0], false, false);
        acc[0][1] = __builtin_amdgcn_wmma_f32_16x16x32_bf16(false, ah0.v, false, b1h.v, (short)0, acc[0][1], false, false);
        acc[1][0] = __builtin_amdgcn_wmma_f32_16x16x32_bf16(false, ah1.v, false, b0h.v, (short)0, acc[1][0], false, false);
        acc[1][1] = __builtin_amdgcn_wmma_f32_16x16x32_bf16(false, ah1.v, false, b1h.v, (short)0, acc[1][1], false, false);
        acc[0][0] = __builtin_amdgcn_wmma_f32_16x16x32_bf16(false, al0.v, false, b0h.v, (short)0, acc[0][0], false, false);
        acc[0][1] = __builtin_amdgcn_wmma_f32_16x16x32_bf16(false, al0.v, false, b1h.v, (short)0, acc[0][1], false, false);
        acc[1][0] = __builtin_amdgcn_wmma_f32_16x16x32_bf16(false, al1.v, false, b0h.v, (short)0, acc[1][0], false, false);
        acc[1][1] = __builtin_amdgcn_wmma_f32_16x16x32_bf16(false, al1.v, false, b1h.v, (short)0, acc[1][1], false, false);
        acc[0][0] = __builtin_amdgcn_wmma_f32_16x16x32_bf16(false, ah0.v, false, b0l.v, (short)0, acc[0][0], false, false);
        acc[0][1] = __builtin_amdgcn_wmma_f32_16x16x32_bf16(false, ah0.v, false, b1l.v, (short)0, acc[0][1], false, false);
        acc[1][0] = __builtin_amdgcn_wmma_f32_16x16x32_bf16(false, ah1.v, false, b0l.v, (short)0, acc[1][0], false, false);
        acc[1][1] = __builtin_amdgcn_wmma_f32_16x16x32_bf16(false, ah1.v, false, b1l.v, (short)0, acc[1][1], false, false);
        __syncthreads();
    }

    // C/D layout: VGPR j of lane l -> M = j + 8*(l>=16), N = l&15
    #pragma unroll
    for (int rt = 0; rt < 2; ++rt) {
        #pragma unroll
        for (int ct = 0; ct < 2; ++ct) {
            v8f a = acc[rt][ct];
            const int nc = nBase + waveCol + (ct << 4) + l15;
            const float bv = (outMode == 0 && bias) ? bias[nc] : 0.0f;
            #pragma unroll
            for (int j = 0; j < 8; ++j) {
                const int mr = mBase + waveRow + (rt << 4) + j + (lhalf << 3);
                const size_t o = (size_t)mr * Ndim + nc;
                if (outMode == 0) {
                    float v = a[j] + bv;
                    if (relu) v = fmaxf(v, 0.0f);
                    C[o] = v;
                } else {
                    float v = a[j];
                    __bf16 h = (__bf16)v;
                    Chi[o] = h;
                    Clo[o] = (__bf16)(v - (float)h);
                }
            }
        }
    }
}

// --------------------------- helper kernels --------------------------------
__global__ void split_bf16(const float* __restrict__ x, __bf16* __restrict__ hi,
                           __bf16* __restrict__ lo, int n)
{
    int i = blockIdx.x * blockDim.x + threadIdx.x;
    if (i < n) {
        float v = x[i];
        __bf16 h = (__bf16)v;
        hi[i] = h;
        lo[i] = (__bf16)(v - (float)h);
    }
}

// split + transpose: x[rows][cols] fp32 -> hiT/loT[cols][rows] bf16
__global__ void splitT_bf16(const float* __restrict__ x, __bf16* __restrict__ hiT,
                            __bf16* __restrict__ loT, int rows, int cols)
{
    int i = blockIdx.x * blockDim.x + threadIdx.x;
    if (i < rows * cols) {
        int r = i / cols, c = i % cols;
        float v = x[i];
        __bf16 h = (__bf16)v;
        size_t o = (size_t)c * rows + r;
        hiT[o] = h;
        loT[o] = (__bf16)(v - (float)h);
    }
}

// WcatT[c][k]: c<128 -> [wt1[c] | wt2[c]], c>=128 -> [wt3[c-128] | wt4[c-128]]
__global__ void build_wcat(const float* __restrict__ w1, const float* __restrict__ w2,
                           const float* __restrict__ w3, const float* __restrict__ w4,
                           const float* __restrict__ b1, const float* __restrict__ b2,
                           const float* __restrict__ b3, const float* __restrict__ b4,
                           __bf16* __restrict__ WhT, __bf16* __restrict__ WlT,
                           float* __restrict__ bcat)
{
    int i = blockIdx.x * blockDim.x + threadIdx.x;   // 256*256
    int c = i >> 8, k = i & 255;
    float v;
    if (c < HID) v = (k < HID) ? w1[c * HID + k] : w2[c * HID + (k - HID)];
    else         v = (k < HID) ? w3[(c - HID) * HID + k] : w4[(c - HID) * HID + (k - HID)];
    __bf16 h = (__bf16)v;
    WhT[i] = h;
    WlT[i] = (__bf16)(v - (float)h);
    if (k == 0) bcat[c] = (c < HID) ? (b1[c] + b2[c]) : (b3[c - HID] + b4[c - HID]);
}

__global__ void lin_sigmoid(const float* __restrict__ x, const float* __restrict__ w,
                            const float* __restrict__ b, float* __restrict__ out,
                            int n, int fin, int fout)
{
    int i = blockIdx.x * blockDim.x + threadIdx.x;
    if (i >= n * fout) return;
    int r = i / fout, j = i % fout;
    const float* xr = x + (size_t)r * fin;
    const float* wj = w + (size_t)j * fin;
    float s = b[j];
    for (int t = 0; t < fin; ++t) s += xr[t] * wj[t];
    out[i] = 1.0f / (1.0f + expf(-s));
}

__global__ void lin_tanh(const float* __restrict__ x, const float* __restrict__ w,
                         const float* __restrict__ b, float* __restrict__ out,
                         int n, int fin, int fout)
{
    int i = blockIdx.x * blockDim.x + threadIdx.x;
    if (i >= n * fout) return;
    int r = i / fout, j = i % fout;
    const float* xr = x + (size_t)r * fin;
    const float* wj = w + (size_t)j * fin;
    float s = b[j];
    for (int t = 0; t < fin; ++t) s += xr[t] * wj[t];
    out[i] = tanhf(s);
}

__global__ void lin_to_state(const float* __restrict__ x, const float* __restrict__ w,
                             const float* __restrict__ b, float* __restrict__ S,
                             int n, int fin, int fout, int colOff, int ldS)
{
    int i = blockIdx.x * blockDim.x + threadIdx.x;
    if (i >= n * fout) return;
    int r = i / fout, j = i % fout;
    const float* xr = x + (size_t)r * fin;
    const float* wj = w + (size_t)j * fin;
    float s = b[j];
    for (int t = 0; t < fin; ++t) s += xr[t] * wj[t];
    S[(size_t)r * ldS + colOff + j] = s;
}

__global__ void sub_kernel(const float* __restrict__ a, const float* __restrict__ b,
                           float* __restrict__ o, int n)
{
    int i = blockIdx.x * blockDim.x + threadIdx.x;
    if (i < n) o[i] = a[i] - b[i];
}

__global__ void heads_kernel(const float* __restrict__ S,
                             const float* __restrict__ w1, const float* __restrict__ bb1,
                             const float* __restrict__ w2, const float* __restrict__ bb2,
                             float* __restrict__ out1, float* __restrict__ out2, int tindex)
{
    int i = blockIdx.x * blockDim.x + threadIdx.x;   // N*4
    if (i >= NROW * 4) return;
    int n = i >> 2, q = i & 3;
    int head = q >> 1, c = q & 1;
    const float* w  = head ? w2 : w1;
    const float* bb = head ? bb2 : bb1;
    const float* sr = S + (size_t)n * CH + head * HID;
    const float* wc = w + c * HID;
    float s = bb[c];
    for (int h = 0; h < HID; ++h) s += sr[h] * wc[h];
    float* o = head ? out2 : out1;
    o[(size_t)tindex * NROW * 2 + n * 2 + c] = 1.0f / (1.0f + expf(-s));
}

// ACC = (init?0:ACC) + accCoef*K;  stage = S + stCoef*dt*K; emit splitT(stage)
// (the fp32 staged state is never materialized -- only its bf16 hi/lo transpose,
//  which is exactly what the next GEMM consumes)
__global__ void rk_axpy_splitT(float* __restrict__ ACC, const float* __restrict__ S,
                               const float* __restrict__ Kv, const float* __restrict__ vt,
                               int step, float accCoef, float stCoef, int init,
                               __bf16* __restrict__ hiT, __bf16* __restrict__ loT)
{
    int i = blockIdx.x * blockDim.x + threadIdx.x;
    if (i >= NROW * CH) return;
    float k = Kv[i];
    float p = init ? 0.0f : ACC[i];
    ACC[i] = p + accCoef * k;
    float dt = vt[step + 1] - vt[step];
    float v = S[i] + stCoef * dt * k;
    int r = i >> 8, c = i & (CH - 1);          // CH == 256
    size_t o = (size_t)c * NROW + r;
    __bf16 h = (__bf16)v;
    hiT[o] = h;
    loT[o] = (__bf16)(v - (float)h);
}

__global__ void acc_op(float* __restrict__ acc, const float* __restrict__ Kv,
                       float coef, int init, int n)
{
    int i = blockIdx.x * blockDim.x + threadIdx.x;
    if (i < n) {
        float p = init ? 0.0f : acc[i];
        acc[i] = p + coef * Kv[i];
    }
}

// S += dt/6 * ACC; emit splitT(S) for the next step's k1 evaluation
__global__ void state_final_splitT(float* __restrict__ S, const float* __restrict__ acc,
                                   const float* __restrict__ vt, int step,
                                   __bf16* __restrict__ hiT, __bf16* __restrict__ loT)
{
    int i = blockIdx.x * blockDim.x + threadIdx.x;
    if (i >= NROW * CH) return;
    float dt = vt[step + 1] - vt[step];
    float v = S[i] + (dt / 6.0f) * acc[i];
    S[i] = v;
    int r = i >> 8, c = i & (CH - 1);
    size_t o = (size_t)c * NROW + r;
    __bf16 h = (__bf16)v;
    hiT[o] = h;
    loT[o] = (__bf16)(v - (float)h);
}

// ---------------------------------------------------------------------------
static inline dim3 g1(int n) { return dim3((n + 255) / 256); }

extern "C" void kernel_launch(void* const* d_in, const int* in_sizes, int n_in,
                              void* d_out, int out_size, void* d_ws, size_t ws_size,
                              hipStream_t stream)
{
    const float* vt     = (const float*)d_in[0];
    const float* A      = (const float*)d_in[1];
    const float* y1     = (const float*)d_in[2];
    const float* y2     = (const float*)d_in[3];
    const float* pre1_w = (const float*)d_in[4];  const float* pre1_b = (const float*)d_in[5];
    const float* pre2_w = (const float*)d_in[6];  const float* pre2_b = (const float*)d_in[7];
    const float* in1_w1 = (const float*)d_in[8];  const float* in1_b1 = (const float*)d_in[9];
    const float* in1_w2 = (const float*)d_in[10]; const float* in1_b2 = (const float*)d_in[11];
    const float* in2_w1 = (const float*)d_in[12]; const float* in2_b1 = (const float*)d_in[13];
    const float* in2_w2 = (const float*)d_in[14]; const float* in2_b2 = (const float*)d_in[15];
    const float* wt1_w  = (const float*)d_in[16]; const float* wt1_b  = (const float*)d_in[17];
    const float* wt2_w  = (const float*)d_in[18]; const float* wt2_b  = (const float*)d_in[19];
    const float* wt3_w  = (const float*)d_in[20]; const float* wt3_b  = (const float*)d_in[21];
    const float* wt4_w  = (const float*)d_in[22]; const float* wt4_b  = (const float*)d_in[23];
    const float* out1_w = (const float*)d_in[24]; const float* out1_b = (const float*)d_in[25];
    const float* out2_w = (const float*)d_in[26]; const float* out2_b = (const float*)d_in[27];

    float* out1 = (float*)d_out;                      // (8, N, 2)
    float* out2 = out1 + (size_t)TSTEPS * NROW * 2;
    float* x0   = out2 + (size_t)TSTEPS * NROW * 2;   // (N, 64)

    // workspace carve-up (~88 MB)
    char* base = (char*)d_ws;
    size_t off = 0;
    auto alloc = [&](size_t bytes) -> void* {
        void* p = base + off;
        off += (bytes + 255) & ~(size_t)255;
        return p;
    };
    __bf16* Ahi  = (__bf16*)alloc((size_t)NROW * NROW * 2);
    __bf16* Alo  = (__bf16*)alloc((size_t)NROW * NROW * 2);
    float*  S    = (float*)alloc((size_t)NROW * CH * 4);
    float*  ACC  = (float*)alloc((size_t)NROW * CH * 4);
    float*  KK   = (float*)alloc((size_t)NROW * CH * 4);
    __bf16* ShiT = (__bf16*)alloc((size_t)CH * NROW * 2);
    __bf16* SloT = (__bf16*)alloc((size_t)CH * NROW * 2);
    __bf16* Ghi  = (__bf16*)alloc((size_t)NROW * CH * 2);
    __bf16* Glo  = (__bf16*)alloc((size_t)NROW * CH * 2);
    __bf16* WhT  = (__bf16*)alloc((size_t)CH * CH * 2);
    __bf16* WlT  = (__bf16*)alloc((size_t)CH * CH * 2);
    float*  bcat = (float*)alloc((size_t)CH * 4);
    float*  h1   = (float*)alloc((size_t)NROW * INDIM * 4);
    float*  h2   = (float*)alloc((size_t)NROW * INDIM * 4);
    float*  tb   = (float*)alloc((size_t)NROW * HID * 4);
    (void)ws_size; (void)in_sizes; (void)n_in; (void)out_size;

    // --- one-time prep: split A, build fused recurrent weights ---
    split_bf16<<<g1(NROW * NROW), 256, 0, stream>>>(A, Ahi, Alo, NROW * NROW);
    build_wcat<<<g1(CH * CH), 256, 0, stream>>>(wt1_w, wt2_w, wt3_w, wt4_w,
                                                wt1_b, wt2_b, wt3_b, wt4_b,
                                                WhT, WlT, bcat);

    // --- front-end MLP: h1/h2, x0, e1/e2 -> state S (N x 256) ---
    lin_sigmoid<<<g1(NROW * INDIM), 256, 0, stream>>>(y1, pre1_w, pre1_b, h1, NROW, INDIM, INDIM);
    lin_sigmoid<<<g1(NROW * INDIM), 256, 0, stream>>>(y2, pre2_w, pre2_b, h2, NROW, INDIM, INDIM);
    sub_kernel<<<g1(NROW * INDIM), 256, 0, stream>>>(h1, h2, x0, NROW * INDIM);
    lin_tanh<<<g1(NROW * HID), 256, 0, stream>>>(h1, in1_w1, in1_b1, tb, NROW, INDIM, HID);
    lin_to_state<<<g1(NROW * HID), 256, 0, stream>>>(tb, in1_w2, in1_b2, S, NROW, HID, HID, 0, CH);
    lin_tanh<<<g1(NROW * HID), 256, 0, stream>>>(h2, in2_w1, in2_b1, tb, NROW, INDIM, HID);
    lin_to_state<<<g1(NROW * HID), 256, 0, stream>>>(tb, in2_w2, in2_b2, S, NROW, HID, HID, HID, CH);
    heads_kernel<<<g1(NROW * 4), 256, 0, stream>>>(S, out1_w, out1_b, out2_w, out2_b, out1, out2, 0);
    splitT_bf16<<<g1(NROW * CH), 256, 0, stream>>>(S, ShiT, SloT, NROW, CH);

    // f(stage): G = A @ stage (WMMA K=4096, split epilogue);
    //           K_out = relu(G @ Wcat + bcat) (WMMA K=256)
    const dim3 gemmGrid(CH / 64, NROW / 128);
    auto runF = [&]() {
        wmma_gemm_bf16x3<<<gemmGrid, 256, 0, stream>>>(
            Ahi, Alo, ShiT, SloT, (const float*)nullptr, (float*)nullptr,
            Ghi, Glo, NROW, CH, NROW, 0, 1);
        wmma_gemm_bf16x3<<<gemmGrid, 256, 0, stream>>>(
            Ghi, Glo, WhT, WlT, bcat, KK,
            (__bf16*)nullptr, (__bf16*)nullptr, NROW, CH, CH, 1, 0);
    };

    const int nstate = NROW * CH;
    for (int step = 0; step < TSTEPS - 1; ++step) {
        runF();   // k1 (from split of S)
        rk_axpy_splitT<<<g1(nstate), 256, 0, stream>>>(ACC, S, KK, vt, step, 1.0f, 0.5f, 1, ShiT, SloT);
        runF();   // k2
        rk_axpy_splitT<<<g1(nstate), 256, 0, stream>>>(ACC, S, KK, vt, step, 2.0f, 0.5f, 0, ShiT, SloT);
        runF();   // k3
        rk_axpy_splitT<<<g1(nstate), 256, 0, stream>>>(ACC, S, KK, vt, step, 2.0f, 1.0f, 0, ShiT, SloT);
        runF();   // k4
        acc_op<<<g1(nstate), 256, 0, stream>>>(ACC, KK, 1.0f, 0, nstate);
        state_final_splitT<<<g1(nstate), 256, 0, stream>>>(S, ACC, vt, step, ShiT, SloT);
        heads_kernel<<<g1(NROW * 4), 256, 0, stream>>>(S, out1_w, out1_b, out2_w, out2_b,
                                                       out1, out2, step + 1);
    }
}
